// Decoder_28621662060864
// MI455X (gfx1250) — matrix operand order repaired and checked by reading, not compiled, not run on hardware
//
#include <hip/hip_runtime.h>
#include <hip/hip_bf16.h>

// ---------------------------------------------------------------------------
// CDNA5 (gfx1250) bf16-WMMA transformer decoder.
//   - all GEMMs + attention einsums via v_wmma_f32_16x16x32_bf16 (wave32)
//   - LDS staging via GLOBAL_LOAD_ASYNC_TO_LDS_B128 (ASYNCcnt), double-buffered
//   - fp32 residual stream, bf16 matmul operands, fp32 accumulation
// ---------------------------------------------------------------------------

typedef __attribute__((ext_vector_type(16))) __bf16 v16bf;
typedef __attribute__((ext_vector_type(8)))  float  v8f;

#define B_     32
#define D_     512
#define NH_    8
#define DH_    64
#define DEPTH_ 8
#define MLP_   2048
#define S_     512        // NC + Hs*Ws
#define NC_    256
#define INNER_ 512
#define M_     (B_ * S_)  // 16384 rows in the token-major activation matrix

static __device__ __forceinline__ unsigned short f2bf(float f) {
    union { float f; unsigned u; } v; v.f = f;
    unsigned u = v.u;
    unsigned r = u + 0x7fffu + ((u >> 16) & 1u);   // round-to-nearest-even
    return (unsigned short)(r >> 16);
}

static __device__ __forceinline__ v8f vzero8() {
    v8f v;
    #pragma unroll
    for (int i = 0; i < 8; ++i) v[i] = 0.0f;
    return v;
}

// 16 bf16 operand fragment, filled from two 16-byte chunks.
union FragBF {
    v16bf v;
    uint4 u[2];
};

// Async global->LDS copy, 16 bytes per lane. lds = 32-bit LDS byte offset
// (low 32 bits of the generic pointer), g = per-lane global byte address.
static __device__ __forceinline__ void async_b128(unsigned lds, unsigned long long g) {
    asm volatile("global_load_async_to_lds_b128 %0, %1, off"
                 :: "v"(lds), "v"(g) : "memory");
}
static __device__ __forceinline__ void wait_async0() {
    asm volatile("s_wait_asynccnt 0" ::: "memory");
}
static __device__ __forceinline__ unsigned lds_off(const void* p) {
    return (unsigned)(unsigned long long)(uintptr_t)p;   // LDS offset = low 32 bits
}

// ---------------------------------------------------------------------------
// Weight convert + transpose: W[z][k][n] (f32) -> Wt[z][n][k] (bf16)
// ---------------------------------------------------------------------------
__global__ void convt_kernel(const float* __restrict__ W,
                             unsigned short* __restrict__ Wt,
                             int K, int N) {
    const size_t mat = (size_t)K * N;
    size_t idx = (size_t)blockIdx.x * blockDim.x + threadIdx.x;
    if (idx >= mat) return;
    const int k = (int)(idx / N);
    const int n = (int)(idx % N);
    const float* Wz = W + (size_t)blockIdx.y * mat;
    unsigned short* Wtz = Wt + (size_t)blockIdx.y * mat;
    Wtz[(size_t)n * K + k] = f2bf(Wz[idx]);
}

// ---------------------------------------------------------------------------
// Build x = concat(slots + slot_pos, z^T + pos)  (raw, pre-norm, fp32)
// ---------------------------------------------------------------------------
__global__ void embed_kernel(const float* __restrict__ z,
                             const float* __restrict__ slots,
                             const float* __restrict__ pos,
                             const float* __restrict__ spos,
                             float* __restrict__ xraw) {
    int idx = blockIdx.x * blockDim.x + threadIdx.x;
    if (idx >= M_ * D_) return;
    const int d = idx & (D_ - 1);
    const int s = (idx >> 9) & (S_ - 1);
    const int b = idx >> 18;
    float v;
    if (s < NC_) {
        v = slots[((size_t)b * NC_ + s) * D_ + d] + spos[(size_t)s * D_ + d];
    } else {
        const int p = s - NC_;
        v = z[((size_t)b * D_ + d) * 256 + p] + pos[(size_t)p * D_ + d];
    }
    xraw[idx] = v;
}

// ---------------------------------------------------------------------------
// LayerNorm over D=512. 128 threads (4 waves), float4 per thread.
// Writes bf16 (yb) and/or fp32 (yf).
// ---------------------------------------------------------------------------
__global__ __launch_bounds__(128)
void ln_kernel(const float* __restrict__ x,
               const float* __restrict__ w,
               const float* __restrict__ b,
               unsigned short* __restrict__ yb,
               float* __restrict__ yf) {
    __shared__ float red[8];
    const int row = blockIdx.x;
    const int tid = threadIdx.x;
    const float4 v = ((const float4*)(x + (size_t)row * D_))[tid];

    float s = v.x + v.y + v.z + v.w;
    #pragma unroll
    for (int o = 1; o < 32; o <<= 1) s += __shfl_xor(s, o, 32);
    if ((tid & 31) == 0) red[tid >> 5] = s;
    __syncthreads();
    const float mean = (red[0] + red[1] + red[2] + red[3]) * (1.0f / (float)D_);

    const float dx = v.x - mean, dy = v.y - mean, dz = v.z - mean, dw = v.w - mean;
    float q = dx * dx + dy * dy + dz * dz + dw * dw;
    #pragma unroll
    for (int o = 1; o < 32; o <<= 1) q += __shfl_xor(q, o, 32);
    if ((tid & 31) == 0) red[4 + (tid >> 5)] = q;
    __syncthreads();
    const float var = (red[4] + red[5] + red[6] + red[7]) * (1.0f / (float)D_);
    const float rs = rsqrtf(var + 1e-5f);

    const int d = tid * 4;
    const float e[4] = { dx, dy, dz, dw };
    #pragma unroll
    for (int j = 0; j < 4; ++j) {
        const float o_ = e[j] * rs * w[d + j] + b[d + j];
        if (yb) yb[(size_t)row * D_ + d + j] = f2bf(o_);
        if (yf) yf[(size_t)row * D_ + d + j] = o_;
    }
}

// ---------------------------------------------------------------------------
// WMMA GEMM: C(MxN) = A(MxK,bf16) * Wt(NxK,bf16)^T  [Wt stored row-major NxK]
// 256 thr = 8 waves. Tile 128x128, BK=32. Wave: 32x64 (2x4 subtiles).
// Async double-buffered LDS staging (GLOBAL_LOAD_ASYNC_TO_LDS_B128).
// mode 0: outB = bf16(acc)
// mode 1: outF += acc + bias[n]        (residual update)
// mode 2: outB = bf16(gelu(acc+bias))  (exact gelu)
// ---------------------------------------------------------------------------
__global__ __launch_bounds__(256)
void gemm_bf16_kernel(const unsigned short* __restrict__ A,
                      const unsigned short* __restrict__ Wt,
                      int M, int N, int K,
                      const float* __restrict__ bias,
                      float* __restrict__ outF,
                      unsigned short* __restrict__ outB,
                      int mode) {
    __shared__ __align__(16) unsigned short As[2][128 * 40];  // 32 + 8 pad
    __shared__ __align__(16) unsigned short Bs[2][128 * 40];

    const int tid  = threadIdx.x;
    const int wave = tid >> 5;
    const int lane = tid & 31;
    const int half = lane >> 4;
    const int lr   = lane & 15;

    const int m0 = blockIdx.y * 128;
    const int n0 = blockIdx.x * 128;
    const int wr = wave >> 1;   // 0..3 : 32-row band
    const int wc = wave & 1;    // 0..1 : 64-col band

    v8f acc[2][4];
    #pragma unroll
    for (int i = 0; i < 2; ++i)
        #pragma unroll
        for (int j = 0; j < 4; ++j) acc[i][j] = vzero8();

    const int srow = tid >> 1;   // 0..127
    const int sh   = tid & 1;    // which 16-elem half of the 32-wide k slice

    // issue async staging for k-block kb into LDS buffer `buf`
    auto issue = [&](int buf, int kb) {
        const unsigned short* ga = A + (size_t)(m0 + srow) * K + kb + sh * 16;
        unsigned la = lds_off(&As[buf][srow * 40 + sh * 16]);
        async_b128(la,      (unsigned long long)(uintptr_t)ga);
        async_b128(la + 16, (unsigned long long)(uintptr_t)(ga + 8));
        const unsigned short* gb = Wt + (size_t)(n0 + srow) * K + kb + sh * 16;
        unsigned lb = lds_off(&Bs[buf][srow * 40 + sh * 16]);
        async_b128(lb,      (unsigned long long)(uintptr_t)gb);
        async_b128(lb + 16, (unsigned long long)(uintptr_t)(gb + 8));
    };

    const int nk = K >> 5;
    issue(0, 0);
    for (int t = 0; t < nk; ++t) {
        wait_async0();
        __syncthreads();
        if (t + 1 < nk) issue((t + 1) & 1, (t + 1) << 5);

        const unsigned short* as = As[t & 1];
        const unsigned short* bs = Bs[t & 1];
        FragBF af[2], bf[4];
        #pragma unroll
        for (int mt = 0; mt < 2; ++mt) {
            const unsigned short* p = as + (wr * 32 + mt * 16 + lr) * 40;
            af[mt].u[0] = *(const uint4*)(p + half * 16);       // k = half*16 + 0..7
            af[mt].u[1] = *(const uint4*)(p + half * 16 + 8);   // k = half*16 + 8..15
        }
        #pragma unroll
        for (int nt = 0; nt < 4; ++nt) {
            const unsigned short* p = bs + (wc * 64 + nt * 16 + lr) * 40 + half * 16;
            bf[nt].u[0] = ((const uint4*)p)[0];                 // k = half*16 + 0..15
            bf[nt].u[1] = ((const uint4*)p)[1];
        }
        #pragma unroll
        for (int mt = 0; mt < 2; ++mt)
            #pragma unroll
            for (int nt = 0; nt < 4; ++nt)
                acc[mt][nt] = __builtin_amdgcn_wmma_f32_16x16x32_bf16(
                    false, af[mt].v, false, bf[nt].v,
                    (short)0, acc[mt][nt], false, false);
    }

    // Epilogue: C layout is lane=N col, vgpr r -> row r + 8*half.
    #pragma unroll
    for (int mt = 0; mt < 2; ++mt) {
        #pragma unroll
        for (int nt = 0; nt < 4; ++nt) {
            const int n = n0 + wc * 64 + nt * 16 + lr;
            #pragma unroll
            for (int r = 0; r < 8; ++r) {
                const int m = m0 + wr * 32 + mt * 16 + r + 8 * half;
                const size_t idx = (size_t)m * N + n;
                const float v = acc[mt][nt][r];
                if (mode == 0) {
                    outB[idx] = f2bf(v);
                } else if (mode == 1) {
                    outF[idx] += v + bias[n];
                } else {
                    const float t = v + bias[n];
                    outB[idx] = f2bf(0.5f * t * (1.0f + erff(t * 0.70710678f)));
                }
            }
        }
    }
}

// ---------------------------------------------------------------------------
// Flash-style causal attention, one (b,h,64-query block) per 128-thr block.
// Scores: A=q(16x64), B=K^T; B-frag wants [n=key][k=d] contiguous in d, so
// row-major K in LDS feeds it directly (K staged with async b128). P*V uses
// V^T in LDS (scalar transpose stores). qkv row (b*S+s), cols [q|k|v].
// ---------------------------------------------------------------------------
__global__ __launch_bounds__(128)
void attn_kernel(const unsigned short* __restrict__ qkv,
                 unsigned short* __restrict__ obuf) {
    __shared__ __align__(16) unsigned short Ks[64 * 72];       // [key][d]
    __shared__ __align__(16) unsigned short Vt[64 * 72];       // [d][key]
    __shared__ __align__(16) unsigned short Ps[4 * 16 * 72];   // per-wave P scratch

    const int tid  = threadIdx.x;
    const int wave = tid >> 5;
    const int lane = tid & 31;
    const int half = lane >> 4;
    const int lr   = lane & 15;

    const int bh = blockIdx.y;
    const int b  = bh >> 3;
    const int h  = bh & (NH_ - 1);
    const int qrow = blockIdx.x * 64 + wave * 16;
    const float scale = 0.125f;  // DH^-0.5

    // q fragments (A layout): two 16x32 k-chunks of the 16x64 q tile
    FragBF qf[2];
    {
        const unsigned short* qp =
            qkv + (size_t)(b * S_ + qrow + lr) * (3 * INNER_) + h * DH_;
        #pragma unroll
        for (int kk = 0; kk < 2; ++kk) {
            qf[kk].u[0] = *(const uint4*)(qp + kk * 32 + half * 16);
            qf[kk].u[1] = *(const uint4*)(qp + kk * 32 + half * 16 + 8);
        }
    }

    v8f O[4];
    #pragma unroll
    for (int i = 0; i < 4; ++i) O[i] = vzero8();
    float mstat[8], lstat[8];
    #pragma unroll
    for (int r = 0; r < 8; ++r) { mstat[r] = -1e30f; lstat[r] = 0.0f; }

    const int nch = blockIdx.x + 1;   // causal: keys up to end of this q block
    for (int kc = 0; kc < nch; ++kc) {
        {   // stage K via async b128 (row-major) and V^T via scalar stores
            const int row = tid >> 1;
            const int hh  = tid & 1;
            const unsigned short* kp =
                qkv + (size_t)(b * S_ + kc * 64 + row) * (3 * INNER_) + INNER_ + h * DH_ + hh * 32;
            unsigned lk = lds_off(&Ks[row * 72 + hh * 32]);
            #pragma unroll
            for (int j = 0; j < 4; ++j)
                async_b128(lk + j * 16, (unsigned long long)(uintptr_t)(kp + j * 8));
            const unsigned short* vp =
                qkv + (size_t)(b * S_ + kc * 64 + row) * (3 * INNER_) + 2 * INNER_ + h * DH_ + hh * 32;
            #pragma unroll
            for (int j = 0; j < 32; ++j)
                Vt[(hh * 32 + j) * 72 + row] = vp[j];
        }
        wait_async0();
        __syncthreads();

        // scores for 4 subtiles of 16 keys (2 WMMA each over DH=64)
        v8f sarr[4];
        #pragma unroll
        for (int nt = 0; nt < 4; ++nt) {
            v8f s = vzero8();
            #pragma unroll
            for (int kk = 0; kk < 2; ++kk) {
                FragBF kf;
                const unsigned short* p = Ks + (nt * 16 + lr) * 72 + kk * 32 + half * 16;
                kf.u[0] = ((const uint4*)p)[0];
                kf.u[1] = ((const uint4*)p)[1];
                s = __builtin_amdgcn_wmma_f32_16x16x32_bf16(
                    false, qf[kk].v, false, kf.v, (short)0, s, false, false);
            }
            const int kt = kc * 64 + nt * 16 + lr;
            #pragma unroll
            for (int r = 0; r < 8; ++r) {
                const int qt = qrow + r + 8 * half;
                s[r] = (kt <= qt) ? s[r] * scale : -1e30f;
            }
            sarr[nt] = s;
        }

        // online softmax (row stats replicated across the 16 lanes of a half)
        float alpha[8];
        #pragma unroll
        for (int r = 0; r < 8; ++r) {
            float rm = fmaxf(fmaxf(sarr[0][r], sarr[1][r]),
                             fmaxf(sarr[2][r], sarr[3][r]));
            #pragma unroll
            for (int o = 1; o < 16; o <<= 1) rm = fmaxf(rm, __shfl_xor(rm, o, 32));
            const float nm = fmaxf(mstat[r], rm);
            alpha[r] = __expf(mstat[r] - nm);
            mstat[r] = nm;
            float rs = 0.0f;
            #pragma unroll
            for (int nt = 0; nt < 4; ++nt) {
                const float p = __expf(sarr[nt][r] - nm);
                sarr[nt][r] = p;
                rs += p;
            }
            #pragma unroll
            for (int o = 1; o < 16; o <<= 1) rs += __shfl_xor(rs, o, 32);
            lstat[r] = lstat[r] * alpha[r] + rs;
        }
        #pragma unroll
        for (int nt = 0; nt < 4; ++nt)
            #pragma unroll
            for (int r = 0; r < 8; ++r)
                O[nt][r] *= alpha[r];

        // P: C layout -> row-major LDS -> reload as A fragments
        unsigned short* pw = Ps + wave * 16 * 72;
        #pragma unroll
        for (int nt = 0; nt < 4; ++nt)
            #pragma unroll
            for (int r = 0; r < 8; ++r)
                pw[(r + 8 * half) * 72 + nt * 16 + lr] = f2bf(sarr[nt][r]);
        __syncthreads();

        #pragma unroll
        for (int kk2 = 0; kk2 < 2; ++kk2) {
            FragBF pf;
            const unsigned short* pp = pw + lr * 72 + kk2 * 32;
            pf.u[0] = *(const uint4*)(pp + half * 16);
            pf.u[1] = *(const uint4*)(pp + half * 16 + 8);
            #pragma unroll
            for (int n2 = 0; n2 < 4; ++n2) {
                FragBF vf;
                const unsigned short* vp2 = Vt + (n2 * 16 + lr) * 72 + kk2 * 32 + half * 16;
                vf.u[0] = ((const uint4*)vp2)[0];
                vf.u[1] = ((const uint4*)vp2)[1];
                O[n2] = __builtin_amdgcn_wmma_f32_16x16x32_bf16(
                    false, pf.v, false, vf.v, (short)0, O[n2], false, false);
            }
        }
        __syncthreads();
    }

    // normalize and store o (bf16, row-major [b*S+s][h*64+d])
    #pragma unroll
    for (int n2 = 0; n2 < 4; ++n2)
        #pragma unroll
        for (int r = 0; r < 8; ++r) {
            const float v = O[n2][r] / lstat[r];
            obuf[(size_t)(b * S_ + qrow + r + 8 * half) * INNER_ + h * DH_ + n2 * 16 + lr] = f2bf(v);
        }
}

// ---------------------------------------------------------------------------
// out[b][d][p] = x[b][NC-1+p][d]
// ---------------------------------------------------------------------------
__global__ void outp_kernel(const float* __restrict__ xf, float* __restrict__ out) {
    int idx = blockIdx.x * blockDim.x + threadIdx.x;
    if (idx >= B_ * D_ * 256) return;
    const int p = idx & 255;
    const int d = (idx >> 8) & (D_ - 1);
    const int b = idx >> 17;
    out[idx] = xf[((size_t)b * S_ + (NC_ - 1 + p)) * D_ + d];
}

// ---------------------------------------------------------------------------
extern "C" void kernel_launch(void* const* d_in, const int* in_sizes, int n_in,
                              void* d_out, int out_size, void* d_ws, size_t ws_size,
                              hipStream_t stream) {
    const float* z        = (const float*)d_in[0];
    const float* slots    = (const float*)d_in[1];
    const float* pos_emb  = (const float*)d_in[2];
    const float* spos_emb = (const float*)d_in[3];
    const float* norm_w   = (const float*)d_in[4];
    const float* norm_b   = (const float*)d_in[5];
    const float* ln1_w    = (const float*)d_in[6];
    const float* ln1_b    = (const float*)d_in[7];
    const float* qkv_w    = (const float*)d_in[8];
    const float* out_w    = (const float*)d_in[9];
    const float* out_b    = (const float*)d_in[10];
    const float* ln2_w    = (const float*)d_in[11];
    const float* ln2_b    = (const float*)d_in[12];
    const float* mlp_w1   = (const float*)d_in[13];
    const float* mlp_b1   = (const float*)d_in[14];
    const float* mlp_w2   = (const float*)d_in[15];
    const float* mlp_b2   = (const float*)d_in[16];
    (void)in_sizes; (void)n_in; (void)out_size; (void)ws_size;

    char* ws = (char*)d_ws;
    size_t off = 0;
    auto alloc = [&](size_t bytes) -> char* {
        char* p = ws + off;
        off += (bytes + 255) & ~(size_t)255;
        return p;
    };
    float* xraw          = (float*)alloc((size_t)M_ * D_ * 4);
    float* xf            = (float*)alloc((size_t)M_ * D_ * 4);
    unsigned short* ybf  = (unsigned short*)alloc((size_t)M_ * D_ * 2);
    unsigned short* qkvb = (unsigned short*)alloc((size_t)M_ * 3 * INNER_ * 2);
    unsigned short* obf  = (unsigned short*)alloc((size_t)M_ * INNER_ * 2);
    unsigned short* h1   = (unsigned short*)alloc((size_t)M_ * MLP_ * 2);
    unsigned short* wqT  = (unsigned short*)alloc((size_t)DEPTH_ * D_ * 3 * INNER_ * 2);
    unsigned short* woT  = (unsigned short*)alloc((size_t)DEPTH_ * INNER_ * D_ * 2);
    unsigned short* w1T  = (unsigned short*)alloc((size_t)DEPTH_ * D_ * MLP_ * 2);
    unsigned short* w2T  = (unsigned short*)alloc((size_t)DEPTH_ * MLP_ * D_ * 2);

    // weights -> bf16 transposed [N][K]
    convt_kernel<<<dim3((D_ * 3 * INNER_ + 255) / 256, DEPTH_), 256, 0, stream>>>(qkv_w, wqT, D_, 3 * INNER_);
    convt_kernel<<<dim3((INNER_ * D_ + 255) / 256, DEPTH_), 256, 0, stream>>>(out_w, woT, INNER_, D_);
    convt_kernel<<<dim3((D_ * MLP_ + 255) / 256, DEPTH_), 256, 0, stream>>>(mlp_w1, w1T, D_, MLP_);
    convt_kernel<<<dim3((MLP_ * D_ + 255) / 256, DEPTH_), 256, 0, stream>>>(mlp_w2, w2T, MLP_, D_);

    embed_kernel<<<(M_ * D_ + 255) / 256, 256, 0, stream>>>(z, slots, pos_emb, spos_emb, xraw);
    ln_kernel<<<M_, 128, 0, stream>>>(xraw, norm_w, norm_b, nullptr, xf);

    for (int i = 0; i < DEPTH_; ++i) {
        ln_kernel<<<M_, 128, 0, stream>>>(xf, ln1_w + i * D_, ln1_b + i * D_, ybf, nullptr);
        gemm_bf16_kernel<<<dim3((3 * INNER_) / 128, M_ / 128), 256, 0, stream>>>(
            ybf, wqT + (size_t)i * D_ * 3 * INNER_, M_, 3 * INNER_, D_,
            nullptr, nullptr, qkvb, 0);
        attn_kernel<<<dim3(S_ / 64, B_ * NH_), 128, 0, stream>>>(qkvb, obf);
        gemm_bf16_kernel<<<dim3(D_ / 128, M_ / 128), 256, 0, stream>>>(
            obf, woT + (size_t)i * INNER_ * D_, M_, D_, INNER_,
            out_b + i * D_, xf, nullptr, 1);
        ln_kernel<<<M_, 128, 0, stream>>>(xf, ln2_w + i * D_, ln2_b + i * D_, ybf, nullptr);
        gemm_bf16_kernel<<<dim3(MLP_ / 128, M_ / 128), 256, 0, stream>>>(
            ybf, w1T + (size_t)i * D_ * MLP_, M_, MLP_, D_,
            mlp_b1 + i * MLP_, nullptr, h1, 2);
        gemm_bf16_kernel<<<dim3(D_ / 128, M_ / 128), 256, 0, stream>>>(
            h1, w2T + (size_t)i * MLP_ * D_, M_, D_, MLP_,
            mlp_b2 + i * D_, xf, nullptr, 1);
    }

    outp_kernel<<<(B_ * D_ * 256 + 255) / 256, 256, 0, stream>>>(xf, (float*)d_out);
}
